// Quantize_60103772340419
// MI455X (gfx1250) — compile-verified, matrix operand order checked by reading, not run on hardware
//
#include <hip/hip_runtime.h>
#include <hip/hip_bf16.h>

typedef __attribute__((ext_vector_type(16))) __bf16 v16bf;
typedef __attribute__((ext_vector_type(8)))  float  v8f;

#define VQ_B 65536
#define VQ_D 512
#define VQ_K 1024

// ---------------------------------------------------------------------------
// Kernel 1: per-code squared norms + bf16 copy of the codebook.
// grid = K blocks x 256 threads; each block handles one code row (512 floats).
// ---------------------------------------------------------------------------
__global__ __launch_bounds__(256) void vq_prep(const float* __restrict__ ew,
                                               float* __restrict__ enorm,
                                               __bf16* __restrict__ ebf) {
    const int k = blockIdx.x;
    const int t = threadIdx.x;
    const float* row = ew + (size_t)k * VQ_D;
    float s = 0.f;
#pragma unroll
    for (int j = t; j < VQ_D; j += 256) {
        float x = row[j];
        s += x * x;
        ebf[(size_t)k * VQ_D + j] = (__bf16)x;
    }
#pragma unroll
    for (int m = 16; m >= 1; m >>= 1) s += __shfl_xor(s, m, 32);
    __shared__ float red[8];
    if ((t & 31) == 0) red[t >> 5] = s;
    __syncthreads();
    if (t == 0) {
        float tot = 0.f;
#pragma unroll
        for (int w = 0; w < 8; ++w) tot += red[w];
        enorm[k] = tot;
    }
}

// ---------------------------------------------------------------------------
// Kernel 2: main VQ kernel. 128 threads = 4 waves; each wave owns 16 rows.
// Distances via bf16 WMMA (f32 accumulate); argmin; gather f32 codebook row;
// accumulate sum((z_q - z)^2) for the losses.
// ---------------------------------------------------------------------------
__global__ __launch_bounds__(128) void vq_main(const float* __restrict__ z,
                                               const float* __restrict__ ew,
                                               const float* __restrict__ enorm,
                                               const __bf16* __restrict__ ebf,
                                               float* __restrict__ out_zq,
                                               float* __restrict__ out_ind,
                                               float* __restrict__ loss_acc) {
    const int wave = threadIdx.x >> 5;
    const int lane = threadIdx.x & 31;
    const long rowbase = ((long)blockIdx.x * 4 + wave) * 16;  // 16 rows per wave
    const int mrow  = lane & 15;          // A layout: lanes 0-15 and 16-31 both map M=0..15
    const int kbase = (lane < 16) ? 0 : 8; // lanes<16: K {0..7,16..23}; lanes>=16: K {8..15,24..31}

    // --- Preload A fragments: 16 chunks of 32 dims, bf16, held in VGPRs ----
    v16bf a[16];
    {
        const float* zr = z + (size_t)(rowbase + mrow) * VQ_D + kbase;
#pragma unroll
        for (int dc = 0; dc < 16; ++dc) {
            const float* p = zr + dc * 32;
            float4 f0 = *(const float4*)(p);
            float4 f1 = *(const float4*)(p + 4);
            float4 f2 = *(const float4*)(p + 16);
            float4 f3 = *(const float4*)(p + 20);
            union { v16bf v; __bf16 e[16]; } u;
            u.e[0]  = (__bf16)f0.x; u.e[1]  = (__bf16)f0.y;
            u.e[2]  = (__bf16)f0.z; u.e[3]  = (__bf16)f0.w;
            u.e[4]  = (__bf16)f1.x; u.e[5]  = (__bf16)f1.y;
            u.e[6]  = (__bf16)f1.z; u.e[7]  = (__bf16)f1.w;
            u.e[8]  = (__bf16)f2.x; u.e[9]  = (__bf16)f2.y;
            u.e[10] = (__bf16)f2.z; u.e[11] = (__bf16)f2.w;
            u.e[12] = (__bf16)f3.x; u.e[13] = (__bf16)f3.y;
            u.e[14] = (__bf16)f3.z; u.e[15] = (__bf16)f3.w;
            a[dc] = u.v;
        }
    }

    // --- Sweep all K=1024 codes in 64 tiles of 16 columns -------------------
    float best[8];
    int   bidx[8];
#pragma unroll
    for (int v = 0; v < 8; ++v) { best[v] = 3.4e38f; bidx[v] = 0; }

    for (int nt = 0; nt < 64; ++nt) {
        const int code = nt * 16 + mrow;   // B layout: lane column = code
        const __bf16* eb = ebf + (size_t)code * VQ_D + kbase;
        v8f acc = {};
#pragma unroll
        for (int dc = 0; dc < 16; ++dc) {
            union { v16bf v; uint4 u[2]; } b;
            b.u[0] = *(const uint4*)(eb + dc * 32);
            b.u[1] = *(const uint4*)(eb + dc * 32 + 16);
            acc = __builtin_amdgcn_wmma_f32_16x16x32_bf16(
                false, a[dc], false, b.v, (short)0, acc, false, false);
        }
        const float en = enorm[code];
#pragma unroll
        for (int v = 0; v < 8; ++v) {
            float d = en - 2.0f * acc[v];   // ||z||^2 is row-constant: drop it
            if (d < best[v]) { best[v] = d; bidx[v] = code; }
        }
    }

    // --- Cross-lane argmin within each 16-lane group ------------------------
#pragma unroll
    for (int v = 0; v < 8; ++v) {
#pragma unroll
        for (int m = 1; m < 16; m <<= 1) {
            float ov = __shfl_xor(best[v], m, 32);
            int   oi = __shfl_xor(bidx[v], m, 32);
            if (ov < best[v] || (ov == best[v] && oi < bidx[v])) {
                best[v] = ov; bidx[v] = oi;
            }
        }
    }
    // group 0 (lanes 0-15): rows v ; group 1 (lanes 16-31): rows 8+v

    // --- Write indices, gather fp32 codebook rows, accumulate loss ----------
    float lsum = 0.f;
#pragma unroll
    for (int v = 0; v < 8; ++v) {
        const int i0 = __shfl(bidx[v], 0, 32);    // row v
        const int i1 = __shfl(bidx[v], 16, 32);   // row 8+v
        if (lane == 0)  out_ind[rowbase + v]     = (float)i0;
        if (lane == 16) out_ind[rowbase + 8 + v] = (float)i1;

        const long r0 = rowbase + v;
        const long r1 = rowbase + 8 + v;
        const int idxs[2] = { i0, i1 };
        const long rows[2] = { r0, r1 };
#pragma unroll
        for (int h = 0; h < 2; ++h) {
            const float4* src = (const float4*)(ew + (size_t)idxs[h] * VQ_D);
            const float4* zs  = (const float4*)(z  + (size_t)rows[h] * VQ_D);
            float4*       dst = (float4*)(out_zq + (size_t)rows[h] * VQ_D);
#pragma unroll
            for (int j = 0; j < 4; ++j) {
                int e = j * 32 + lane;          // 128 float4 per row / 32 lanes
                float4 q  = src[e];
                float4 zz = zs[e];
                dst[e] = q;                     // z_q_st == z_q value-wise
                float dx = q.x - zz.x, dy = q.y - zz.y;
                float dz = q.z - zz.z, dw = q.w - zz.w;
                lsum += dx * dx + dy * dy + dz * dz + dw * dw;
            }
        }
    }
#pragma unroll
    for (int m = 16; m >= 1; m >>= 1) lsum += __shfl_xor(lsum, m, 32);
    if (lane == 0) atomicAdd(loss_acc, lsum);
}

// ---------------------------------------------------------------------------
// Kernel 3: finalize the two loss scalars.
// ---------------------------------------------------------------------------
__global__ void vq_final(const float* __restrict__ loss_acc,
                         float* __restrict__ out_losses) {
    const float s   = *loss_acc;
    const float inv = 1.0f / ((float)VQ_B * (float)VQ_D);
    out_losses[0] = 2.5f  * s * inv;   // commit:   0.25 * 10 * mean
    out_losses[1] = 10.0f * s * inv;   // codebook: 10 * mean
}

extern "C" void kernel_launch(void* const* d_in, const int* in_sizes, int n_in,
                              void* d_out, int out_size, void* d_ws, size_t ws_size,
                              hipStream_t stream) {
    const float* z  = (const float*)d_in[0];   // [B, D]
    const float* ew = (const float*)d_in[1];   // [K, D]

    float* out    = (float*)d_out;
    float* zq     = out;                               // B*D floats
    float* losses = out + (size_t)VQ_B * VQ_D;         // 2 floats
    float* indf   = losses + 2;                        // B floats (argmin indices)

    unsigned char* ws = (unsigned char*)d_ws;
    float*  enorm = (float*)ws;                                    // 4 KB
    __bf16* ebf   = (__bf16*)(ws + 4096);                          // 1 MB bf16 codebook
    float*  lacc  = (float*)(ws + 4096 + (size_t)VQ_K * VQ_D * 2); // 1 float

    hipMemsetAsync(lacc, 0, sizeof(float), stream);
    vq_prep<<<VQ_K, 256, 0, stream>>>(ew, enorm, ebf);
    vq_main<<<VQ_B / 64, 128, 0, stream>>>(z, ew, enorm, ebf, zq, indf, lacc);
    vq_final<<<1, 1, 0, stream>>>(lacc, losses);
}
